// CurloraLayer_5196910428211
// MI455X (gfx1250) — compile-verified, matrix operand order checked by reading, not run on hardware
//
#include <hip/hip_runtime.h>
#include <hip/hip_bf16.h>

// ---------------------------------------------------------------------------
// CURLoRA forward on MI455X (gfx1250):
//   out = x @ (W + C@U@R)^T + bias
// Strategy:
//   1) CU = C@U                                  (tiny)
//   2) W' = W + CU@R, split into bf16 hi/lo      (workspace, 64 MB)
//   3) out = x @ W'^T via v_wmma_f32_16x16x32_bf16 with 3-product hi/lo
//      split (fp32-faithful), 128x128 block tiles, double-buffered LDS,
//      W' tiles staged with GLOBAL_LOAD_ASYNC_TO_LDS (ASYNCcnt path),
//      non-temporal output stores to preserve L2 for W'/X reuse.
// ---------------------------------------------------------------------------

typedef __attribute__((ext_vector_type(16))) __bf16 v16bf;
typedef __attribute__((ext_vector_type(8)))  __bf16 bf16x8;
typedef __attribute__((ext_vector_type(4)))  __bf16 bf16x4;
typedef __attribute__((ext_vector_type(8)))  float  v8f;
typedef __attribute__((ext_vector_type(4)))  int    v4i;

#define M_TOT 8192   // B*S
#define N_TOT 4096   // D_OUT
#define K_TOT 4096   // D_IN
#define RANK  64
#define NKT   (K_TOT / 32)

#ifndef __has_builtin
#define __has_builtin(x) 0
#endif

#if __has_builtin(__builtin_amdgcn_global_load_async_to_lds_b128)
#define HAVE_ASYNC_LDS 1
#else
#define HAVE_ASYNC_LDS 0
#endif
#if __has_builtin(__builtin_amdgcn_s_wait_asynccnt)
#define HAVE_WAIT_ASYNC 1
#else
#define HAVE_WAIT_ASYNC 0
#endif

#define AS1 __attribute__((address_space(1)))
#define AS3 __attribute__((address_space(3)))

// ---------------- Kernel 1: CU = C @ U  ([4096,64] = [4096,64]x[64,64]) ----
__global__ __launch_bounds__(256) void cu_kernel(const float* __restrict__ C,
                                                 const float* __restrict__ U,
                                                 float* __restrict__ CU) {
  int idx = blockIdx.x * 256 + threadIdx.x;     // 4096*64 threads
  int r = idx & (RANK - 1);
  int o = idx >> 6;
  float acc = 0.0f;
#pragma unroll
  for (int j = 0; j < RANK; ++j)
    acc = fmaf(C[o * RANK + j], U[j * RANK + r], acc);
  CU[idx] = acc;
}

// ---------------- Kernel 2: W' = W + CU@R, split to bf16 hi/lo -------------
__global__ __launch_bounds__(256) void wprime_kernel(const float* __restrict__ W,
                                                     const float* __restrict__ CU,
                                                     const float* __restrict__ R,
                                                     __bf16* __restrict__ Whi,
                                                     __bf16* __restrict__ Wlo) {
  long idx = (long)blockIdx.x * 256 + threadIdx.x;  // 4096*4096 threads
  int d = (int)(idx & (K_TOT - 1));                 // coalesced along d
  int o = (int)(idx >> 12);
  float acc = W[idx];
#pragma unroll 16
  for (int r = 0; r < RANK; ++r)
    acc = fmaf(CU[o * RANK + r], R[r * K_TOT + d], acc);
  __bf16 hi = (__bf16)acc;                 // RNE to bf16
  __bf16 lo = (__bf16)(acc - (float)hi);   // residual in bf16
  Whi[idx] = hi;
  Wlo[idx] = lo;
}

// Load a 16x32 bf16 fragment row slice per the CDNA5 16-bit A/B VGPR layout:
// lanes 0-15 hold K chunks {0..7, 16..23}, lanes 16-31 hold {8..15, 24..31}.
__device__ __forceinline__ v16bf ldFrag(const __bf16* rowPtr, int b0) {
  union { v16bf v; bf16x8 h[2]; } t;
  t.h[0] = *(const bf16x8*)(rowPtr + b0);
  t.h[1] = *(const bf16x8*)(rowPtr + b0 + 16);
  return t.v;
}

__device__ __forceinline__ void waitAsync() {
#if HAVE_ASYNC_LDS
#if HAVE_WAIT_ASYNC
  __builtin_amdgcn_s_wait_asynccnt(0);
#else
  asm volatile("s_wait_asynccnt 0x0" ::: "memory");
#endif
#endif
}

// One 16-byte async global->LDS copy (GLOBAL_LOAD_ASYNC_TO_LDS_B128).
__device__ __forceinline__ void asyncCopy16(const __bf16* g, __bf16* l) {
#if HAVE_ASYNC_LDS
  // Convert pointee type in generic AS first, then pure addrspace conversion.
  v4i* gp = (v4i*)(void*)const_cast<__bf16*>(g);
  v4i* lp = (v4i*)(void*)l;
  __builtin_amdgcn_global_load_async_to_lds_b128((AS1 v4i*)gp, (AS3 v4i*)lp, 0, 0);
#else
  *(bf16x8*)l = *(const bf16x8*)g;
#endif
}

// Stage one 128(N) x 32(K) tile of each W' plane into LDS.
__device__ __forceinline__ void stageW(const __bf16* __restrict__ Whi,
                                       const __bf16* __restrict__ Wlo,
                                       __bf16* swhi, __bf16* swlo,
                                       int nBase, int kBase, int wr, int wk) {
#pragma unroll
  for (int p = 0; p < 2; ++p) {
    const int row = p * 64 + wr;
    const long g  = (long)(nBase + row) * K_TOT + kBase + wk;
    const int  l  = row * 32 + wk;
    asyncCopy16(Whi + g, swhi + l);
    asyncCopy16(Wlo + g, swlo + l);
  }
}

// Stage one 128(M) x 32(K) f32 tile of X, split to bf16 hi/lo, into LDS.
__device__ __forceinline__ void stageX(const float* __restrict__ X,
                                       __bf16* sxhi, __bf16* sxlo,
                                       int mBase, int kBase, int xr, int xk) {
#pragma unroll
  for (int p = 0; p < 4; ++p) {
    const int row = p * 32 + xr;
    const float4 v =
        *(const float4*)(X + (long)(mBase + row) * K_TOT + kBase + xk);
    // High-locality prefetch of the next K tile -> WGP-scope prefetch that
    // pulls into all cache levels (global_prefetch_b8, SCOPE_WGP).
    __builtin_prefetch(X + (long)(mBase + row) * K_TOT + kBase + 32 + xk, 0, 3);
    const __bf16 h0 = (__bf16)v.x, h1 = (__bf16)v.y,
                 h2 = (__bf16)v.z, h3 = (__bf16)v.w;
    const bf16x4 hv = {h0, h1, h2, h3};
    const bf16x4 lv = {(__bf16)(v.x - (float)h0), (__bf16)(v.y - (float)h1),
                       (__bf16)(v.z - (float)h2), (__bf16)(v.w - (float)h3)};
    *(bf16x4*)(sxhi + row * 32 + xk) = hv;
    *(bf16x4*)(sxlo + row * 32 + xk) = lv;
  }
}

// ---------------- Kernel 3: main GEMM, out = X @ W'^T + bias ---------------
// Block: 256 threads = 8 waves (wave32). Block tile 128(M) x 128(N), K step
// 32, double-buffered LDS. Wave grid 4(M) x 2(N): each wave owns 32x64 =
// 2x4 fragments of 16x16 -> 24 v_wmma per K step (hi/lo 3-product split).
__global__ __launch_bounds__(256) void gemm_kernel(const float* __restrict__ X,
                                                   const __bf16* __restrict__ Whi,
                                                   const __bf16* __restrict__ Wlo,
                                                   const float* __restrict__ bias,
                                                   float* __restrict__ out) {
  __shared__ __bf16 sXhi[2][128 * 32];
  __shared__ __bf16 sXlo[2][128 * 32];
  __shared__ __bf16 sWhi[2][128 * 32];
  __shared__ __bf16 sWlo[2][128 * 32];

  const int tid   = threadIdx.x;
  const int lane  = tid & 31;
  const int wave  = tid >> 5;
  const int waveM = wave >> 1;   // 0..3
  const int waveN = wave & 1;    // 0..1

  const int mBase = blockIdx.y * 128;
  const int nBase = blockIdx.x * 128;

  v8f acc[2][4];
#pragma unroll
  for (int mf = 0; mf < 2; ++mf)
#pragma unroll
    for (int nf = 0; nf < 4; ++nf)
#pragma unroll
      for (int i = 0; i < 8; ++i) acc[mf][nf][i] = 0.0f;

  const int lrow = lane & 15;          // row/col within a 16-wide fragment
  const int b0   = (lane >> 4) * 8;    // K-chunk select per half-wave

  // X staging geometry: one float4 per thread, 8 threads/row, 4 passes.
  const int xr = tid >> 3;             // 0..31
  const int xk = (tid & 7) * 4;        // 0,4,...,28
  // W staging geometry: one 16B b128 per plane, 4 threads/row, 2 passes.
  const int wr = tid >> 2;             // 0..63
  const int wk = (tid & 3) * 8;        // 0,8,16,24

  // ---- prologue: stage K-tile 0 into buffer 0 ----
  stageW(Whi, Wlo, sWhi[0], sWlo[0], nBase, 0, wr, wk);
  stageX(X, sXhi[0], sXlo[0], mBase, 0, xr, xk);
  waitAsync();
  __syncthreads();

  for (int kt = 0; kt < NKT; ++kt) {
    const int buf = kt & 1;

    // ---- fragment loads from the current buffer (issue early) ----
    v16bf ahi[2], alo[2], bhi[4], blo[4];
#pragma unroll
    for (int mf = 0; mf < 2; ++mf) {
      const int row = waveM * 32 + mf * 16 + lrow;
      ahi[mf] = ldFrag(sXhi[buf] + row * 32, b0);
      alo[mf] = ldFrag(sXlo[buf] + row * 32, b0);
    }
#pragma unroll
    for (int nf = 0; nf < 4; ++nf) {
      const int row = waveN * 64 + nf * 16 + lrow;
      bhi[nf] = ldFrag(sWhi[buf] + row * 32, b0);
      blo[nf] = ldFrag(sWlo[buf] + row * 32, b0);
    }

    // ---- stage next K-tile into the other buffer (overlaps with WMMA) ----
    if (kt + 1 < NKT) {
      const int nb = buf ^ 1;
      const int kNext = (kt + 1) * 32;
      stageW(Whi, Wlo, sWhi[nb], sWlo[nb], nBase, kNext, wr, wk);
      stageX(X, sXhi[nb], sXlo[nb], mBase, kNext, xr, xk);
    }

    // ---- 3-product hi/lo WMMA accumulation (drop lo*lo) ----
#pragma unroll
    for (int mf = 0; mf < 2; ++mf)
#pragma unroll
      for (int nf = 0; nf < 4; ++nf) {
        acc[mf][nf] = __builtin_amdgcn_wmma_f32_16x16x32_bf16(
            false, ahi[mf], false, bhi[nf], (short)0, acc[mf][nf], false, false);
        acc[mf][nf] = __builtin_amdgcn_wmma_f32_16x16x32_bf16(
            false, ahi[mf], false, blo[nf], (short)0, acc[mf][nf], false, false);
        acc[mf][nf] = __builtin_amdgcn_wmma_f32_16x16x32_bf16(
            false, alo[mf], false, bhi[nf], (short)0, acc[mf][nf], false, false);
      }

    waitAsync();       // next buffer's async W copies complete
    __syncthreads();   // next buffer visible to all waves; current buffer free
  }

  // ---- epilogue: bias add + non-temporal store (16x16 f32 C/D layout) ----
#pragma unroll
  for (int nf = 0; nf < 4; ++nf) {
    const int col = nBase + waveN * 64 + nf * 16 + lrow;
    const float bv = bias[col];
#pragma unroll
    for (int mf = 0; mf < 2; ++mf) {
      const int rbase = mBase + waveM * 32 + mf * 16 + (lane >> 4) * 8;
#pragma unroll
      for (int i = 0; i < 8; ++i)
        __builtin_nontemporal_store(acc[mf][nf][i] + bv,
                                    out + (long)(rbase + i) * N_TOT + col);
    }
  }
}

// ---------------------------------------------------------------------------
extern "C" void kernel_launch(void* const* d_in, const int* in_sizes, int n_in,
                              void* d_out, int out_size, void* d_ws, size_t ws_size,
                              hipStream_t stream) {
  const float* x    = (const float*)d_in[0];  // [4,2048,4096]
  const float* W    = (const float*)d_in[1];  // [4096,4096]
  const float* C    = (const float*)d_in[2];  // [4096,64]
  const float* U    = (const float*)d_in[3];  // [64,64]
  const float* R    = (const float*)d_in[4];  // [64,4096]
  const float* bias = (const float*)d_in[5];  // [4096]
  float* out = (float*)d_out;                 // [4,2048,4096]

  char* ws = (char*)d_ws;
  float*  CU  = (float*)ws;                                   // 1 MB
  __bf16* Whi = (__bf16*)(ws + (1u << 20));                   // 32 MB
  __bf16* Wlo = (__bf16*)(ws + (1u << 20) + (32u << 20));     // 32 MB

  cu_kernel<<<(N_TOT * RANK) / 256, 256, 0, stream>>>(C, U, CU);
  wprime_kernel<<<((long)N_TOT * K_TOT) / 256, 256, 0, stream>>>(W, CU, R, Whi, Wlo);

  dim3 grid(N_TOT / 128, M_TOT / 128);  // (32, 64)
  gemm_kernel<<<grid, 256, 0, stream>>>(x, Whi, Wlo, bias, out);
}
